// RelPartialLearnableMultiHeadAttn_31645319037264
// MI455X (gfx1250) — matrix operand order repaired and checked by reading, not compiled, not run on hardware
//
#include <hip/hip_runtime.h>
#include <hip/hip_bf16.h>

#define SEQL 1024
#define MEMN 1024
#define BATCH 4
#define HDIM 1024
#define NHEAD 16
#define HD 64
#define TOT 2048
#define RLEN 2048
#define RKP 2080            // RLEN + 32 zero-pad rows for rel-shift overrun
#define ATT_SCALE 0.125f    // 1/sqrt(64)
#define LN_EPS 1e-5f

typedef __bf16 bf16;
typedef __attribute__((ext_vector_type(16))) __bf16 v16bf;
typedef __attribute__((ext_vector_type(8)))  float  v8f;
typedef unsigned int v4u __attribute__((ext_vector_type(4)));
typedef int          v8i __attribute__((ext_vector_type(8)));
typedef int          v4i __attribute__((ext_vector_type(4)));

// ---------------------------------------------------------------------------
// WMMA helper: D = A(16x32 bf16) * B(32x16 bf16) + C(16x16 f32)
// ---------------------------------------------------------------------------
__device__ __forceinline__ v8f wmma_bf16(v16bf a, v16bf b, v8f c) {
  return __builtin_amdgcn_wmma_f32_16x16x32_bf16(false, a, false, b,
                                                 (short)0, c, false, false);
}

// A-operand fragment (16x32, 16-bit): lane L<16 holds M=L, K in {0..7,16..23};
// lane L>=16 holds M=L-16, K in {8..15,24..31}.  klo = (lane&16)?8:0.
__device__ __forceinline__ v16bf a_frag_bf16(const bf16* __restrict__ row, int klo) {
  v16bf a;
#pragma unroll
  for (int s = 0; s < 8; ++s) a[s] = row[klo + s];
#pragma unroll
  for (int s = 0; s < 8; ++s) a[s + 8] = row[klo + 16 + s];
  return a;
}
__device__ __forceinline__ v16bf a_frag_f32(const float* __restrict__ row, int klo) {
  v16bf a;
#pragma unroll
  for (int s = 0; s < 8; ++s) a[s] = (bf16)row[klo + s];
#pragma unroll
  for (int s = 0; s < 8; ++s) a[s + 8] = (bf16)row[klo + 16 + s];
  return a;
}

// B-operand fragment (32x16, 16-bit): lane L holds N=L&15; K in {klo..klo+7,
// klo+16..klo+23}, klo = koff/2, koff = (lane&16)?16:0.  rowN points at the K-major row for column N.
__device__ __forceinline__ v16bf b_frag_bf16(const bf16* __restrict__ rowN, int koff) {
  v16bf b;
#pragma unroll
  for (int s = 0; s < 16; ++s) b[s] = rowN[(koff >> 1) + (s & 7) + ((s >> 3) << 4)];
  return b;
}

// ---------------------------------------------------------------------------
// Tensor Data Mover: async 2-D bf16 tile load (global -> LDS), D# per
// CDNA5 ISA.  count=1, type=2, data_size=2B,
// no gather/iterate/pad, groups 2/3 zero (2-D tensor).
// 6-arg builtin form (clang-23 / therock-10.0 headers).
// ---------------------------------------------------------------------------
__device__ __forceinline__ void tdm_load_2d(unsigned lds_addr, const void* gaddr,
                                            unsigned tensor_d0, unsigned tensor_d1,
                                            unsigned tile_d0, unsigned tile_d1,
                                            unsigned stride0_elems) {
  unsigned long long ga = (unsigned long long)gaddr;
  v4u g0;
  g0.x = 1u;                                         // count=1, user descriptor
  g0.y = lds_addr;                                   // LDS byte address
  g0.z = (unsigned)(ga & 0xFFFFFFFFu);               // global addr [31:0]
  g0.w = (unsigned)((ga >> 32) & 0x1FFFFFFu) | (2u << 30);  // [56:32] | type=2
  v8i g1;
  g1[0] = (int)(1u << 16);                           // data_size = 2 bytes
  g1[1] = (int)((tensor_d0 & 0xFFFFu) << 16);        // tensor_dim0[15:0] @63:48
  g1[2] = (int)(((tensor_d0 >> 16) & 0xFFFFu) |      // tensor_dim0[31:16]
                ((tensor_d1 & 0xFFFFu) << 16));      // tensor_dim1[15:0]
  g1[3] = (int)(((tensor_d1 >> 16) & 0xFFFFu) |      // tensor_dim1[31:16]
                ((tile_d0 & 0xFFFFu) << 16));        // tile_dim0
  g1[4] = (int)(tile_d1 & 0xFFFFu);                  // tile_dim1, tile_dim2=0
  g1[5] = (int)stride0_elems;                        // tensor_dim0_stride[31:0]
  g1[6] = 0;                                         // stride0 hi / stride1 lo
  g1[7] = 0;
  v4i g2 = {0, 0, 0, 0};
  v4i g3 = {0, 0, 0, 0};
  v8i g4 = {0, 0, 0, 0, 0, 0, 0, 0};
  __builtin_amdgcn_tensor_load_to_lds(g0, g1, g2, g3, g4, 0);
}

// ---------------------------------------------------------------------------
// Prep: f32 -> bf16 conversions (one-time; strips cvt VALU out of GEMM loops)
// ---------------------------------------------------------------------------
__global__ __launch_bounds__(256) void cvt_f32_bf16(
    const float* __restrict__ src, bf16* __restrict__ dst, int n) {
  int i = blockIdx.x * 256 + threadIdx.x;
  if (i < n) dst[i] = (bf16)src[i];
}

__global__ __launch_bounds__(256) void pack_full_bf16(
    const float* __restrict__ x, const float* __restrict__ memv,
    bf16* __restrict__ full) {
  int i = blockIdx.x * 256 + threadIdx.x;     // over TOT*B*H
  if (i < TOT * BATCH * HDIM) {
    const int row = i / HDIM;                 // t*B + b
    const int t = row >> 2;
    full[i] = (bf16)((t < MEMN) ? memv[i] : x[i - MEMN * BATCH * HDIM]);
  }
}

__global__ void rk_pad_zero(bf16* __restrict__ Rk) {
  int idx = blockIdx.x * blockDim.x + threadIdx.x;
  const int total = NHEAD * (RKP - RLEN) * HD;
  if (idx < total) {
    int h = idx / ((RKP - RLEN) * HD);
    int rem = idx % ((RKP - RLEN) * HD);
    int p = RLEN + rem / HD;
    int d = rem % HD;
    Rk[((size_t)h * RKP + p) * HD + d] = (bf16)0.0f;
  }
}

// ---------------------------------------------------------------------------
// QKV projection: full (TOT*B, H) @ W_qkv^T, 16x64 per wave, scatter into
// Qw = q + r_w_bias, Qr = q + r_r_bias [b][h][i][d]; K [b][h][j][d];
// V^T [b][h][d][j]  (all bf16)
// ---------------------------------------------------------------------------
__global__ __launch_bounds__(32) void qkv_gemm(
    const bf16* __restrict__ fullb, const bf16* __restrict__ Wqkvb,
    const float* __restrict__ rwb, const float* __restrict__ rrb,
    bf16* __restrict__ Qw, bf16* __restrict__ Qr,
    bf16* __restrict__ Kb, bf16* __restrict__ Vt) {
  const int lane = threadIdx.x;
  const int ln = lane & 15;
  const int klo = (lane & 16) ? 8 : 0;
  const int koff = (lane & 16) ? 16 : 0;
  const int hi8 = (lane >> 4) << 3;
  const int mtile = blockIdx.x;          // 512
  const int ntile = blockIdx.y;          // 48

  const bf16* arow = fullb + (size_t)(mtile * 16 + ln) * HDIM;
  const bf16* brow[4];
#pragma unroll
  for (int f = 0; f < 4; ++f)
    brow[f] = Wqkvb + (size_t)(ntile * 64 + f * 16 + ln) * HDIM;

  v8f acc[4] = {v8f{}, v8f{}, v8f{}, v8f{}};
  for (int kb = 0; kb < HDIM; kb += 32) {
    __builtin_prefetch(arow + kb + 256, 0, 1);
    v16bf a = a_frag_bf16(arow + kb, klo);
#pragma unroll
    for (int f = 0; f < 4; ++f)
      acc[f] = wmma_bf16(a, b_frag_bf16(brow[f] + kb, koff), acc[f]);
  }

#pragma unroll
  for (int f = 0; f < 4; ++f) {
    const int n = ntile * 64 + f * 16 + ln;
#pragma unroll
    for (int r = 0; r < 8; ++r) {
      const int m = mtile * 16 + hi8 + r;
      const int tt = m >> 2, b2 = m & 3;
      const float v = acc[f][r];
      if (n < HDIM) {                 // Q (only current-segment rows)
        if (tt >= MEMN) {
          const int i = tt - MEMN, h = n >> 6, d = n & 63;
          const size_t q = (((size_t)b2 * NHEAD + h) * SEQL + i) * HD + d;
          Qw[q] = (bf16)(v + rwb[n]);
          Qr[q] = (bf16)(v + rrb[n]);
        }
      } else if (n < 2 * HDIM) {      // K
        const int nn = n - HDIM, h = nn >> 6, d = nn & 63;
        Kb[(((size_t)b2 * NHEAD + h) * TOT + tt) * HD + d] = (bf16)v;
      } else {                        // V transposed
        const int nn = n - 2 * HDIM, h = nn >> 6, d = nn & 63;
        Vt[(((size_t)b2 * NHEAD + h) * HD + d) * TOT + tt] = (bf16)v;
      }
    }
  }
}

// ---------------------------------------------------------------------------
// R projection: pos_embed (RLEN,H) @ W_r^T -> Rk [h][p][d]  (bf16)
// ---------------------------------------------------------------------------
__global__ __launch_bounds__(32) void r_gemm(
    const bf16* __restrict__ posb, const bf16* __restrict__ Wrb,
    bf16* __restrict__ Rk) {
  const int lane = threadIdx.x;
  const int ln = lane & 15;
  const int klo = (lane & 16) ? 8 : 0;
  const int koff = (lane & 16) ? 16 : 0;
  const int hi8 = (lane >> 4) << 3;
  const int mtile = blockIdx.x;          // 128
  const int ntile = blockIdx.y;          // 16

  const bf16* arow = posb + (size_t)(mtile * 16 + ln) * HDIM;
  const bf16* brow[4];
#pragma unroll
  for (int f = 0; f < 4; ++f)
    brow[f] = Wrb + (size_t)(ntile * 64 + f * 16 + ln) * HDIM;

  v8f acc[4] = {v8f{}, v8f{}, v8f{}, v8f{}};
  for (int kb = 0; kb < HDIM; kb += 32) {
    v16bf a = a_frag_bf16(arow + kb, klo);
#pragma unroll
    for (int f = 0; f < 4; ++f)
      acc[f] = wmma_bf16(a, b_frag_bf16(brow[f] + kb, koff), acc[f]);
  }
#pragma unroll
  for (int f = 0; f < 4; ++f) {
    const int n = ntile * 64 + f * 16 + ln;
    const int h = n >> 6, d = n & 63;
#pragma unroll
    for (int r = 0; r < 8; ++r) {
      const int p = mtile * 16 + hi8 + r;
      Rk[((size_t)h * RKP + p) * HD + d] = (bf16)acc[f][r];
    }
  }
}

// ---------------------------------------------------------------------------
// Fused rel-attention, flash style, TDM double-buffered K/V staging in LDS.
// One wave handles 16 query rows of one (b,h). Per 32-key step:
// 4 WMMA AC + 6 WMMA BD (48 shifted cols via LDS) + 4 WMMA P*V.
// BD[i,j] = (q_i + r_r_bias) . r_k[j - i + SEQL - 1]  (rel-shift identity)
// ---------------------------------------------------------------------------
__global__ __launch_bounds__(32) void attn_flash(
    const bf16* __restrict__ Qw, const bf16* __restrict__ Qr,
    const bf16* __restrict__ Kb, const bf16* __restrict__ Vt,
    const bf16* __restrict__ Rk, bf16* __restrict__ avec) {
  __shared__ bf16 ldsK[2][32 * HD];     // [buf][j][d]
  __shared__ bf16 ldsV[2][HD * 32];     // [buf][d][j]
  __shared__ float ldsS[16 * 48];
  const int lane = threadIdx.x;
  const int ln = lane & 15;
  const int klo = (lane & 16) ? 8 : 0;
  const int koff = (lane & 16) ? 16 : 0;
  const int hi8 = (lane >> 4) << 3;
  const int i0 = blockIdx.x * 16;
  const int h = blockIdx.y;
  const int b = blockIdx.z;
  const size_t bh = (size_t)b * NHEAD + h;

  const bf16* qwrow = Qw + (bh * SEQL + i0 + ln) * HD;
  const bf16* qrrow = Qr + (bh * SEQL + i0 + ln) * HD;
  const v16bf aqw0 = a_frag_bf16(qwrow, klo);
  const v16bf aqw1 = a_frag_bf16(qwrow + 32, klo);
  const v16bf aqr0 = a_frag_bf16(qrrow, klo);
  const v16bf aqr1 = a_frag_bf16(qrrow + 32, klo);

  const bf16* kbase = Kb + bh * TOT * HD;
  const bf16* vbase = Vt + bh * HD * TOT;

  // prologue: async-stage first K / V^T tiles via the Tensor Data Mover
  tdm_load_2d((unsigned)(size_t)&ldsK[0][0], kbase, HD, TOT, HD, 32, HD);
  tdm_load_2d((unsigned)(size_t)&ldsV[0][0], vbase, TOT, HD, 32, HD, TOT);

  v8f o[4] = {v8f{}, v8f{}, v8f{}, v8f{}};
  float rowmax[8], rowsum[8];
#pragma unroll
  for (int r = 0; r < 8; ++r) { rowmax[r] = -3.0e38f; rowsum[r] = 0.0f; }

  const int kvlen = i0 + 16 + MEMN;
  int it = 0;
  for (int jp = 0; jp < kvlen; jp += 32, ++it) {
    const int buf = it & 1;
    // prefetch next tiles into the other buffer (clamped; extra load harmless)
    int jn = jp + 32;
    if (jn > TOT - 32) jn = TOT - 32;
    asm volatile("s_wait_dscnt 0x0" ::: "memory");  // LDS reads done before DMA overwrite
    tdm_load_2d((unsigned)(size_t)&ldsK[buf ^ 1][0], kbase + (size_t)jn * HD,
                HD, TOT - jn, HD, 32, HD);
    tdm_load_2d((unsigned)(size_t)&ldsV[buf ^ 1][0], vbase + jn,
                TOT - jn, HD, 32, HD, TOT);
    __builtin_amdgcn_s_wait_tensorcnt(2);           // current buffer landed

    const bf16* kt = &ldsK[buf][0];
    const bf16* vt = &ldsV[buf][0];

    // ---- AC = (q+r_w_bias) . K^T, two 16-key subtiles -------------------
    v8f ac[2];
#pragma unroll
    for (int tsub = 0; tsub < 2; ++tsub) {
      const bf16* krow = kt + (tsub * 16 + ln) * HD;
      v8f acc = {};
      acc = wmma_bf16(aqw0, b_frag_bf16(krow, koff), acc);
      acc = wmma_bf16(aqw1, b_frag_bf16(krow + 32, koff), acc);
      ac[tsub] = acc;
    }
    // ---- BD pre-shift: 3 column tiles covering 48 shifted columns -------
    const int base = jp - i0 + (SEQL - 16);  // >= 0, +47 <= RKP-1
#pragma unroll
    for (int u = 0; u < 3; ++u) {
      const bf16* rrow = Rk + ((size_t)h * RKP + (base + u * 16 + ln)) * HD;
      v8f bd = {};
      bd = wmma_bf16(aqr0, b_frag_bf16(rrow, koff), bd);
      bd = wmma_bf16(aqr1, b_frag_bf16(rrow + 32, koff), bd);
#pragma unroll
      for (int r = 0; r < 8; ++r) ldsS[(hi8 + r) * 48 + u * 16 + ln] = bd[r];
    }
    // ---- scores + online softmax ---------------------------------------
    float p0[8], p1[8], sf[8];
#pragma unroll
    for (int r = 0; r < 8; ++r) {
      const int M = hi8 + r;
      float s0 = ac[0][r] + ldsS[M * 48 + (ln - M + 15)];
      float s1 = ac[1][r] + ldsS[M * 48 + (ln - M + 31)];
      s0 *= ATT_SCALE;
      s1 *= ATT_SCALE;
      const int lim = i0 + M + MEMN;         // mask: j > i + MEM
      if (jp + ln > lim)      s0 = -3.0e38f;
      if (jp + 16 + ln > lim) s1 = -3.0e38f;
      float vmax = fmaxf(s0, s1);
#pragma unroll
      for (int m = 1; m < 16; m <<= 1)
        vmax = fmaxf(vmax, __shfl_xor(vmax, m, 32));
      const float newm = fmaxf(rowmax[r], vmax);
      const float scl = __expf(rowmax[r] - newm);
      const float e0 = __expf(s0 - newm);
      const float e1 = __expf(s1 - newm);
      float ps = e0 + e1;
#pragma unroll
      for (int m = 1; m < 16; m <<= 1)
        ps += __shfl_xor(ps, m, 32);
      rowsum[r] = rowsum[r] * scl + ps;
      rowmax[r] = newm;
      sf[r] = scl;
      p0[r] = e0;
      p1[r] = e1;
    }
    // ---- stage P in LDS (BD reads complete), rescale O ------------------
#pragma unroll
    for (int r = 0; r < 8; ++r) {
      const int M = hi8 + r;
      ldsS[M * 48 + ln] = p0[r];
      ldsS[M * 48 + 16 + ln] = p1[r];
    }
#pragma unroll
    for (int c = 0; c < 4; ++c)
#pragma unroll
      for (int r = 0; r < 8; ++r) o[c][r] *= sf[r];
    // ---- P (16x32) * V (32x64) via 4 WMMAs, V^T from LDS ----------------
    const v16bf pf = a_frag_f32(ldsS + ln * 48, klo);
#pragma unroll
    for (int c = 0; c < 4; ++c) {
      const bf16* vrow = vt + (c * 16 + ln) * 32;
      o[c] = wmma_bf16(pf, b_frag_bf16(vrow, koff), o[c]);
    }
  }
  // ---- normalize and emit attn_vec (bf16) for the output GEMM -----------
#pragma unroll
  for (int c = 0; c < 4; ++c)
#pragma unroll
    for (int r = 0; r < 8; ++r) {
      const int M = hi8 + r;
      const float v = o[c][r] / rowsum[r];
      const size_t idx = ((size_t)(i0 + M) * BATCH + b) * (NHEAD * HD)
                       + (size_t)h * HD + c * 16 + ln;
      avec[idx] = (bf16)v;
    }
}

// ---------------------------------------------------------------------------
// Output projection: attn_vec (SEQ*B, 1024) @ W_o^T -> attn_out f32
// ---------------------------------------------------------------------------
__global__ __launch_bounds__(32) void out_gemm(
    const bf16* __restrict__ avec, const bf16* __restrict__ Wob,
    float* __restrict__ aout) {
  const int lane = threadIdx.x;
  const int ln = lane & 15;
  const int klo = (lane & 16) ? 8 : 0;
  const int koff = (lane & 16) ? 16 : 0;
  const int hi8 = (lane >> 4) << 3;
  const int mtile = blockIdx.x;          // 256
  const int ntile = blockIdx.y;          // 16

  const bf16* arow = avec + (size_t)(mtile * 16 + ln) * HDIM;
  const bf16* brow[4];
#pragma unroll
  for (int f = 0; f < 4; ++f)
    brow[f] = Wob + (size_t)(ntile * 64 + f * 16 + ln) * HDIM;

  v8f acc[4] = {v8f{}, v8f{}, v8f{}, v8f{}};
  for (int kb = 0; kb < HDIM; kb += 32) {
    v16bf a = a_frag_bf16(arow + kb, klo);
#pragma unroll
    for (int f = 0; f < 4; ++f)
      acc[f] = wmma_bf16(a, b_frag_bf16(brow[f] + kb, koff), acc[f]);
  }
#pragma unroll
  for (int f = 0; f < 4; ++f) {
    const int n = ntile * 64 + f * 16 + ln;
#pragma unroll
    for (int r = 0; r < 8; ++r) {
      const size_t m = (size_t)mtile * 16 + hi8 + r;
      aout[m * HDIM + n] = acc[f][r];
    }
  }
}

// ---------------------------------------------------------------------------
// Residual + LayerNorm: out = LN(x + attn_out) * w + b
// ---------------------------------------------------------------------------
__global__ __launch_bounds__(256) void ln_kernel(
    const float* __restrict__ x, const float* __restrict__ aout,
    const float* __restrict__ w, const float* __restrict__ bias,
    float* __restrict__ out) {
  __shared__ float red[256];
  const int row = blockIdx.x;
  const int tid = threadIdx.x;
  const float* xr = x + (size_t)row * HDIM;
  const float* ar = aout + (size_t)row * HDIM;
  float vals[4];
  float s = 0.0f;
#pragma unroll
  for (int q = 0; q < 4; ++q) {
    vals[q] = xr[tid + q * 256] + ar[tid + q * 256];
    s += vals[q];
  }
  red[tid] = s;
  __syncthreads();
  for (int off = 128; off > 0; off >>= 1) {
    if (tid < off) red[tid] += red[tid + off];
    __syncthreads();
  }
  const float mu = red[0] * (1.0f / HDIM);
  __syncthreads();
  float vs = 0.0f;
#pragma unroll
  for (int q = 0; q < 4; ++q) {
    const float d = vals[q] - mu;
    vs += d * d;
  }
  red[tid] = vs;
  __syncthreads();
  for (int off = 128; off > 0; off >>= 1) {
    if (tid < off) red[tid] += red[tid + off];
    __syncthreads();
  }
  const float rstd = rsqrtf(red[0] * (1.0f / HDIM) + LN_EPS);
#pragma unroll
  for (int q = 0; q < 4; ++q) {
    const int c = tid + q * 256;
    out[(size_t)row * HDIM + c] = (vals[q] - mu) * rstd * w[c] + bias[c];
  }
}

// ---------------------------------------------------------------------------
extern "C" void kernel_launch(void* const* d_in, const int* in_sizes, int n_in,
                              void* d_out, int out_size, void* d_ws,
                              size_t ws_size, hipStream_t stream) {
  const float* x    = (const float*)d_in[0];
  const float* pos  = (const float*)d_in[1];
  const float* rwb  = (const float*)d_in[2];
  const float* rrb  = (const float*)d_in[3];
  /* d_in[4] attn_mask: causal mask computed analytically in-kernel */
  const float* memv = (const float*)d_in[5];
  const float* Wqkv = (const float*)d_in[6];
  const float* Wr   = (const float*)d_in[7];
  const float* Wo   = (const float*)d_in[8];
  const float* lnw  = (const float*)d_in[9];
  const float* lnb  = (const float*)d_in[10];
  float* out = (float*)d_out;

  char* ws = (char*)d_ws;
  size_t off = 0;
  auto alloc = [&](size_t bytes) {
    char* p = ws + off;
    off += (bytes + 255) & ~(size_t)255;
    return p;
  };
  bf16* fullb = (bf16*)alloc((size_t)TOT * BATCH * HDIM * 2);     // 16 MB
  bf16* Wqkvb = (bf16*)alloc((size_t)3 * HDIM * HDIM * 2);        // 6 MB
  bf16* Wrb   = (bf16*)alloc((size_t)HDIM * HDIM * 2);            // 2 MB
  bf16* Wob   = (bf16*)alloc((size_t)HDIM * HDIM * 2);            // 2 MB
  bf16* posb  = (bf16*)alloc((size_t)RLEN * HDIM * 2);            // 4 MB
  bf16* Qw    = (bf16*)alloc((size_t)BATCH * NHEAD * SEQL * HD * 2);
  bf16* Qr    = (bf16*)alloc((size_t)BATCH * NHEAD * SEQL * HD * 2);
  bf16* Kb    = (bf16*)alloc((size_t)BATCH * NHEAD * TOT * HD * 2);
  bf16* Vt    = (bf16*)alloc((size_t)BATCH * NHEAD * TOT * HD * 2);
  bf16* Rk    = (bf16*)alloc((size_t)NHEAD * RKP * HD * 2);
  bf16* avec  = (bf16*)alloc((size_t)SEQL * BATCH * HDIM * 2);
  float* aout = (float*)alloc((size_t)SEQL * BATCH * HDIM * 4);

  // prep: one-time f32 -> bf16 conversions
  pack_full_bf16<<<(TOT * BATCH * HDIM) / 256, 256, 0, stream>>>(x, memv, fullb);
  cvt_f32_bf16<<<(3 * HDIM * HDIM) / 256, 256, 0, stream>>>(Wqkv, Wqkvb,
                                                            3 * HDIM * HDIM);
  cvt_f32_bf16<<<(HDIM * HDIM) / 256, 256, 0, stream>>>(Wr, Wrb, HDIM * HDIM);
  cvt_f32_bf16<<<(HDIM * HDIM) / 256, 256, 0, stream>>>(Wo, Wob, HDIM * HDIM);
  cvt_f32_bf16<<<(RLEN * HDIM) / 256, 256, 0, stream>>>(pos, posb, RLEN * HDIM);
  rk_pad_zero<<<(NHEAD * (RKP - RLEN) * HD + 255) / 256, 256, 0, stream>>>(Rk);

  qkv_gemm<<<dim3((TOT * BATCH) / 16, (3 * HDIM) / 64), 32, 0, stream>>>(
      fullb, Wqkvb, rwb, rrb, Qw, Qr, Kb, Vt);
  r_gemm<<<dim3(RLEN / 16, HDIM / 64), 32, 0, stream>>>(posb, Wrb, Rk);
  attn_flash<<<dim3(SEQL / 16, NHEAD, BATCH), 32, 0, stream>>>(
      Qw, Qr, Kb, Vt, Rk, avec);
  out_gemm<<<dim3((SEQL * BATCH) / 16, HDIM / 64), 32, 0, stream>>>(
      avec, Wob, aout);
  ln_kernel<<<SEQL * BATCH, 256, 0, stream>>>(x, aout, lnw, lnb, out);
}
